// VisualTransformer_16423954940145
// MI455X (gfx1250) — matrix operand order repaired and checked by reading, not compiled, not run on hardware
//
#include <hip/hip_runtime.h>
#include <math.h>

// ============================================================================
// VisualTransformer forward for MI455X (gfx1250, wave32, WMMA).
// S=1024 B=2 D=1024 H=16 HD=64 MLP=4096 L=4.
//
// All GEMMs run on v_wmma_f32_16x16x32_bf16 with every matrix operand
// materialized as bf16 in memory (weights staged per layer; LN / RoPE /
// softmax / GEMM-epilogue emit bf16 directly), so the GEMM inner loop is
// pure global_load_b128 + v_wmma.
//
// GEMM tiling: block 256 threads = 8 waves, block tile 256x64, wave tile
// 32x64 (2 A fragments x 4 B fragments = 8 WMMA per k-step, B fragments
// reused across both A fragments and across the 8 waves via cache).
// The k-loop is software-pipelined with double-buffered fragments so loads
// for step i+1 are in flight during the WMMAs of step i (avoids the
// s_wait_loadcnt 0x0 full drain before every WMMA), plus global_prefetch
// for the A stream.
//
// fp32 kept only where accuracy matters: residual stream, pre-RoPE qkv,
// raw attention scores. Workspace use: ~224 MB (fp32 scores dominate).
// ============================================================================

typedef unsigned short u16;
typedef __attribute__((ext_vector_type(16))) __bf16 v16bf;
typedef __attribute__((ext_vector_type(8)))  float  v8f;

__device__ __forceinline__ u16 bf16r(float f) {
    // round-to-nearest-even fp32 -> bf16 (producers only, never in GEMM loop)
    unsigned u = __float_as_uint(f);
    unsigned r = u + 0x7FFFu + ((u >> 16) & 1u);
    return (u16)(r >> 16);
}
__device__ __forceinline__ float bf2f(u16 h) {
    return __uint_as_float(((unsigned)h) << 16);
}

union ABF { v16bf v; uint4 q[2]; };

// ----------------------------------------------------------------------------
// fp32 -> bf16 bulk convert (n % 4 == 0). One-time per layer for weights.
// ----------------------------------------------------------------------------
__global__ void __launch_bounds__(256)
cvt_bf16(const float* __restrict__ x, u16* __restrict__ y, long n)
{
    long i4 = (long)blockIdx.x * 256 + threadIdx.x;
    if (i4 * 4 >= n) return;
    float4 f = ((const float4*)x)[i4];
    uint2 o;
    o.x = (unsigned)bf16r(f.x) | ((unsigned)bf16r(f.y) << 16);
    o.y = (unsigned)bf16r(f.z) | ((unsigned)bf16r(f.w) << 16);
    ((uint2*)y)[i4] = o;
}

// ----------------------------------------------------------------------------
// Batched WMMA GEMM: C[z,m,n] = alpha * sum_k A[z,m,k] * W[z,n,k]
//                     (+ bias[n]) (+ res) (GELU) ; bf16 A/W, fp32 or bf16 C.
// A: bf16 row-major MxK (lda in elements), W: bf16 row-major NxK (ldw).
// flags: bit0 = exact GELU, bit1 = residual add (fp32, C-index), bit2 = C bf16.
// Requires M%256==0, N%64==0, K%64==0; lda/ldw element offsets 8-aligned.
// ----------------------------------------------------------------------------
__global__ void __launch_bounds__(256)
wmma_gemm(const u16* __restrict__ A, const u16* __restrict__ W,
          const float* __restrict__ bias, const float* __restrict__ res,
          void* __restrict__ Cv,
          int M, int N, int K, int lda, int ldw, int ldc,
          long aB, long wB, long cB, float alpha, int flags)
{
    const int tid  = threadIdx.x;
    const int w    = tid >> 5;        // wave id 0..7
    const int lane = tid & 31;
    const int half = lane >> 4;       // lanes 0-15 vs 16-31
    const int nl   = lane & 15;
    const int z    = blockIdx.z;

    const int rowA0   = blockIdx.y * 256 + w * 32 + nl;   // A fragment 0 rows
    const int colBase = blockIdx.x * 64;

    // Per-lane stream pointers.
    const u16* pa0 = A + (size_t)z * aB + (size_t)rowA0 * lda;
    const u16* pa1 = pa0 + (size_t)16 * lda;              // A fragment 1 rows
    const u16* pwb = W + (size_t)z * wB;
    const u16* pw[4];
#pragma unroll
    for (int j = 0; j < 4; j++)
        pw[j] = pwb + (size_t)(colBase + j * 16 + nl) * ldw;

    v8f acc[2][4] = {};
    ABF aF[2][2];   // [buf][m-subtile]
    ABF bF[2][4];   // [buf][n-subtile]

    // A fragment (16x32 bf16, ISA interleaved layout): lanes 0-15 hold
    // K {0..7} then {16..23}; lanes 16-31 shifted by +8 => two 16B loads.
    // B fragment (32x16): same interleaved K layout as A => two 16B loads.
#define LD_FRAGS(buf, k)                                                     \
    do {                                                                     \
        aF[buf][0].q[0] = *(const uint4*)(pa0 + (k) + 8 * half);             \
        aF[buf][0].q[1] = *(const uint4*)(pa0 + (k) + 16 + 8 * half);        \
        aF[buf][1].q[0] = *(const uint4*)(pa1 + (k) + 8 * half);             \
        aF[buf][1].q[1] = *(const uint4*)(pa1 + (k) + 16 + 8 * half);        \
        _Pragma("unroll")                                                    \
        for (int j = 0; j < 4; j++) {                                        \
            bF[buf][j].q[0] = *(const uint4*)(pw[j] + (k) + 8 * half);       \
            bF[buf][j].q[1] = *(const uint4*)(pw[j] + (k) + 16 + 8 * half);  \
        }                                                                    \
    } while (0)

#define MM_FRAGS(buf)                                                        \
    do {                                                                     \
        _Pragma("unroll")                                                    \
        for (int j = 0; j < 4; j++)                                          \
            acc[0][j] = __builtin_amdgcn_wmma_f32_16x16x32_bf16(             \
                false, aF[buf][0].v, false, bF[buf][j].v, (short)0,          \
                acc[0][j], false, false);                                    \
        _Pragma("unroll")                                                    \
        for (int j = 0; j < 4; j++)                                          \
            acc[1][j] = __builtin_amdgcn_wmma_f32_16x16x32_bf16(             \
                false, aF[buf][1].v, false, bF[buf][j].v, (short)0,          \
                acc[1][j], false, false);                                    \
    } while (0)

    LD_FRAGS(0, 0);
    for (int kk = 0; kk < K; kk += 64) {
        // loads for the odd k-step fly under the even k-step's WMMAs
        LD_FRAGS(1, kk + 32);
        __builtin_prefetch(pa0 + kk + 256, 0, 1);   // global_prefetch (A rows)
        __builtin_prefetch(pa1 + kk + 256, 0, 1);
        MM_FRAGS(0);
        int kb = (kk + 64 < K) ? kk + 64 : 0;       // clamp: last iter reloads k=0
        LD_FRAGS(0, kb);
        MM_FRAGS(1);
    }
#undef LD_FRAGS
#undef MM_FRAGS

    const bool do_gelu = (flags & 1) != 0;
    const bool do_res  = (flags & 2) != 0;
    const bool out_bf  = (flags & 4) != 0;
#pragma unroll
    for (int mi = 0; mi < 2; mi++) {
#pragma unroll
        for (int j = 0; j < 4; j++) {
            int col = colBase + j * 16 + nl;
#pragma unroll
            for (int r = 0; r < 8; r++) {
                int row = blockIdx.y * 256 + w * 32 + mi * 16 + half * 8 + r;
                float v = acc[mi][j][r] * alpha;
                if (bias) v += bias[col];
                size_t ci = (size_t)z * cB + (size_t)row * ldc + col;
                if (do_res) v += res[ci];
                if (do_gelu) v = 0.5f * v * (1.0f + erff(v * 0.70710678118654752f));
                if (out_bf) ((u16*)Cv)[ci] = bf16r(v);
                else        ((float*)Cv)[ci] = v;
            }
        }
    }
}

// ----------------------------------------------------------------------------
// LayerNorm over last dim (D=1024), fp32 in; fp32 or bf16 out.
// ----------------------------------------------------------------------------
__global__ void __launch_bounds__(256)
ln_kernel(const float* __restrict__ x, const float* __restrict__ g,
          const float* __restrict__ b, void* __restrict__ y, int R, int out_bf)
{
    __shared__ float red[256];
    int row = blockIdx.x;
    if (row >= R) return;
    const float* xr = x + (size_t)row * 1024;
    float s = 0.f, ss = 0.f;
    for (int c = threadIdx.x; c < 1024; c += 256) {
        float v = xr[c]; s += v; ss += v * v;
    }
    red[threadIdx.x] = s; __syncthreads();
    for (int st = 128; st > 0; st >>= 1) {
        if (threadIdx.x < st) red[threadIdx.x] += red[threadIdx.x + st];
        __syncthreads();
    }
    float mean = red[0] * (1.f / 1024.f);
    __syncthreads();
    red[threadIdx.x] = ss; __syncthreads();
    for (int st = 128; st > 0; st >>= 1) {
        if (threadIdx.x < st) red[threadIdx.x] += red[threadIdx.x + st];
        __syncthreads();
    }
    float var  = red[0] * (1.f / 1024.f) - mean * mean;
    float rstd = rsqrtf(var + 1e-5f);
    for (int c = threadIdx.x; c < 1024; c += 256) {
        float v = (xr[c] - mean) * rstd * g[c] + b[c];
        size_t i = (size_t)row * 1024 + c;
        if (out_bf) ((u16*)y)[i] = bf16r(v);
        else        ((float*)y)[i] = v;
    }
}

// ----------------------------------------------------------------------------
// Split qkv (S,B,3,H,HD) fp32 -> bf16 Q,K with RoPE in (B,H,S,HD);
// V transposed to (B,H,HD,S) so the PV GEMM B-fragment loads are contiguous.
// tid = ((b*H+h)*S + s)*HD + d
// ----------------------------------------------------------------------------
__global__ void __launch_bounds__(256)
rope_split(const float* __restrict__ qkv, const float* __restrict__ rpe,
           u16* __restrict__ Q, u16* __restrict__ Ko, u16* __restrict__ Vt)
{
    int tid = blockIdx.x * blockDim.x + threadIdx.x;
    const int total = 2 * 16 * 1024 * 64;
    if (tid >= total) return;
    int d = tid & 63;
    int s = (tid >> 6) & 1023;
    int h = (tid >> 16) & 15;
    int b = tid >> 20;

    size_t base = ((size_t)(s * 2 + b)) * 3072;   // qkv row (s,b), 3*D wide
    float ang = rpe[s * 32 + (d & 31)];
    float cv = cosf(ang), sv = sinf(ang);

    float qv = qkv[base + h * 64 + d];
    float kv = qkv[base + 1024 + h * 64 + d];
    float vv = qkv[base + 2048 + h * 64 + d];
    int pd  = (d < 32) ? d + 32 : d - 32;
    float sgn = (d < 32) ? -1.f : 1.f;
    float qp = qkv[base + h * 64 + pd];
    float kp = qkv[base + 1024 + h * 64 + pd];

    size_t oi = (size_t)tid;                       // (B,H,S,HD) flat
    Q[oi]  = bf16r(qv * cv + sgn * qp * sv);
    Ko[oi] = bf16r(kv * cv + sgn * kp * sv);
    Vt[(((size_t)(b * 16 + h) * 64 + d) * 1024) + s] = bf16r(vv);
}

// ----------------------------------------------------------------------------
// Row softmax on fp32 scores, result written IN PLACE as bf16 at the same
// row base (bf16 row occupies first half of the fp32 row's bytes; each row
// is private to one block; all fp32 reads complete before the block's
// barriers, bf16 writes happen after the last barrier -> no aliasing hazard).
// Row length fixed at 1024 (4 values/thread, kept in registers).
// ----------------------------------------------------------------------------
__global__ void __launch_bounds__(256)
softmax_rows_bf16(float* __restrict__ sc)
{
    __shared__ float red[256];
    float* row = sc + (size_t)blockIdx.x * 1024;
    u16* orow = (u16*)row;

    float v[4];
    float mx = -3.4e38f;
#pragma unroll
    for (int i = 0; i < 4; i++) {
        v[i] = row[threadIdx.x + i * 256];
        mx = fmaxf(mx, v[i]);
    }
    red[threadIdx.x] = mx; __syncthreads();
    for (int st = 128; st > 0; st >>= 1) {
        if (threadIdx.x < st)
            red[threadIdx.x] = fmaxf(red[threadIdx.x], red[threadIdx.x + st]);
        __syncthreads();
    }
    mx = red[0]; __syncthreads();
    float s = 0.f;
#pragma unroll
    for (int i = 0; i < 4; i++) { v[i] = __expf(v[i] - mx); s += v[i]; }
    red[threadIdx.x] = s; __syncthreads();
    for (int st = 128; st > 0; st >>= 1) {
        if (threadIdx.x < st) red[threadIdx.x] += red[threadIdx.x + st];
        __syncthreads();
    }
    float inv = 1.f / red[0];
#pragma unroll
    for (int i = 0; i < 4; i++)
        orow[threadIdx.x + i * 256] = bf16r(v[i] * inv);
}

// ----------------------------------------------------------------------------
// Pooling head helpers (tiny FLOPs -> scalar kernels).
// ----------------------------------------------------------------------------
__global__ void meanpool(const float* __restrict__ h, float* __restrict__ xq)
{
    int tid = blockIdx.x * blockDim.x + threadIdx.x;
    if (tid >= 2048) return;
    int b = tid >> 10, d = tid & 1023;
    float s = 0.f;
    for (int t = 0; t < 1024; t++) s += h[((size_t)t * 2 + b) * 1024 + d];
    xq[tid] = s * (1.f / 1024.f);
}

__global__ void rowmat(const float* __restrict__ X, const float* __restrict__ W,
                       const float* __restrict__ bias, float* __restrict__ Y,
                       int R, int N, int K)
{
    int tid = blockIdx.x * blockDim.x + threadIdx.x;
    if (tid >= R * N) return;
    int r = tid / N, e = tid % N;
    const float* x = X + (size_t)r * K;
    const float* w = W + (size_t)e * K;
    float s = 0.f;
    for (int k = 0; k < K; k++) s += x[k] * w[k];
    if (bias) s += bias[e];
    Y[tid] = s;
}

// One block per (b,h): single-query cross attention over 1024 tokens.
// k,v bf16 in (S,B,D) layout; q,o fp32 in (B,D).
__global__ void __launch_bounds__(256)
pool_attn(const float* __restrict__ q, const u16* __restrict__ k,
          const u16* __restrict__ v, float* __restrict__ o)
{
    __shared__ float p[1024];
    __shared__ float red[256];
    __shared__ float sq[64];
    int bh = blockIdx.x, b = bh >> 4, h = bh & 15;
    if (threadIdx.x < 64) sq[threadIdx.x] = q[b * 1024 + h * 64 + threadIdx.x];
    __syncthreads();
    for (int t = threadIdx.x; t < 1024; t += 256) {
        const u16* kr = k + ((size_t)t * 2 + b) * 1024 + h * 64;
        float s = 0.f;
        for (int d = 0; d < 64; d++) s += sq[d] * bf2f(kr[d]);
        p[t] = s * 0.125f;
    }
    __syncthreads();
    float mx = -3.4e38f;
    for (int t = threadIdx.x; t < 1024; t += 256) mx = fmaxf(mx, p[t]);
    red[threadIdx.x] = mx; __syncthreads();
    for (int st = 128; st > 0; st >>= 1) {
        if (threadIdx.x < st)
            red[threadIdx.x] = fmaxf(red[threadIdx.x], red[threadIdx.x + st]);
        __syncthreads();
    }
    mx = red[0]; __syncthreads();
    float s = 0.f;
    for (int t = threadIdx.x; t < 1024; t += 256) {
        float e = __expf(p[t] - mx); p[t] = e; s += e;
    }
    red[threadIdx.x] = s; __syncthreads();
    for (int st = 128; st > 0; st >>= 1) {
        if (threadIdx.x < st) red[threadIdx.x] += red[threadIdx.x + st];
        __syncthreads();
    }
    float inv = 1.f / red[0];
    __syncthreads();
    if (threadIdx.x < 64) {
        int d = threadIdx.x;
        float acc = 0.f;
        for (int t = 0; t < 1024; t++)
            acc += p[t] * bf2f(v[((size_t)t * 2 + b) * 1024 + h * 64 + d]);
        o[b * 1024 + h * 64 + d] = acc * inv;
    }
}

// ----------------------------------------------------------------------------
static inline void launch_gemm(const u16* A, const u16* W, const float* bias,
                               const float* res, void* C,
                               int M, int N, int K, int lda, int ldw, int ldc,
                               long aB, long wB, long cB, int batches,
                               float alpha, int flags, hipStream_t s)
{
    dim3 grid(N / 64, M / 256, batches), block(256);
    wmma_gemm<<<grid, block, 0, s>>>(A, W, bias, res, C, M, N, K,
                                     lda, ldw, ldc, aB, wB, cB, alpha, flags);
}

static inline void launch_cvt(const float* x, u16* y, long n, hipStream_t s)
{
    dim3 grid((unsigned)((n / 4 + 255) / 256)), block(256);
    cvt_bf16<<<grid, block, 0, s>>>(x, y, n);
}

extern "C" void kernel_launch(void* const* d_in, const int* in_sizes, int n_in,
                              void* d_out, int out_size, void* d_ws, size_t ws_size,
                              hipStream_t stream)
{
    constexpr int S = 1024, B = 2, D = 1024, H = 16, HD = 64, MLPD = 4096, L = 4;
    constexpr int M2 = S * B;  // 2048 token rows

    const float* x    = (const float*)d_in[0];
    const float* rpe  = (const float*)d_in[1];
    const float* ln1w = (const float*)d_in[2];
    const float* ln1b = (const float*)d_in[3];
    const float* inw  = (const float*)d_in[4];
    const float* inb  = (const float*)d_in[5];
    const float* outw = (const float*)d_in[6];
    const float* outb = (const float*)d_in[7];
    const float* ln2w = (const float*)d_in[8];
    const float* ln2b = (const float*)d_in[9];
    const float* fcw  = (const float*)d_in[10];
    const float* fcb  = (const float*)d_in[11];
    const float* cpw  = (const float*)d_in[12];
    const float* cpb  = (const float*)d_in[13];
    const float* pnqw = (const float*)d_in[14];
    const float* pnqb = (const float*)d_in[15];
    const float* pnkw = (const float*)d_in[16];
    const float* pnkb = (const float*)d_in[17];
    const float* pnvw = (const float*)d_in[18];
    const float* pnvb = (const float*)d_in[19];
    const float* pqw  = (const float*)d_in[20];
    const float* pkw  = (const float*)d_in[21];
    const float* pvw  = (const float*)d_in[22];
    const float* ppw  = (const float*)d_in[23];
    const float* ppb  = (const float*)d_in[24];

    // ---- Workspace layout -------------------------------------------
    // fp32 region: 41M floats (164 MB); bf16 region: 30M halves (60 MB).
    float* ws  = (float*)d_ws;
    float* h   = ws;                         // 2M floats  residual (S,B,D)
    float* qkv = ws + (2ull  << 20);         // 6M floats  (S,B,3D)
    float* sc  = ws + (8ull  << 20);         // 32M floats (B,H,S,S) scores
    float* xq  = ws + (40ull << 20);         // small fp32 pooling temps
    float* xqn = xq  + 2048;
    float* qp  = xqn + 2048;
    float* op  = qp  + 2048;

    u16* ub   = (u16*)(ws + (41ull << 20));
    u16* an   = ub;                          // 2M  LN out / xk (bf16)
    u16* Qb   = ub + (2ull  << 20);          // 2M  Q (B,H,S,HD) / kpool
    u16* Kb   = ub + (4ull  << 20);          // 2M  K (B,H,S,HD) / vpool
    u16* Vtb  = ub + (6ull  << 20);          // 2M  V^T (B,H,HD,S)
    u16* ob   = ub + (8ull  << 20);          // 2M  attn out (S,B,D)
    u16* midb = ub + (10ull << 20);          // 8M  MLP hidden / xv
    u16* wl   = ub + (18ull << 20);          // 12M per-layer bf16 weights
    u16* wl_in = wl;                         // 3M
    u16* wl_ou = wl + (3ull << 20);          // 1M
    u16* wl_fc = wl + (4ull << 20);          // 4M
    u16* wl_cp = wl + (8ull << 20);          // 4M

    const float scale = 0.125f;  // 1/sqrt(HD)

    // h = x
    hipMemcpyAsync(h, x, sizeof(float) * (size_t)M2 * D,
                   hipMemcpyDeviceToDevice, stream);

    for (int l = 0; l < L; l++) {
        // ---- stage this layer's weights as bf16 ---------------------
        launch_cvt(inw  + (size_t)l * 3 * D * D, wl_in, 3L * D * D, stream);
        launch_cvt(outw + (size_t)l * D * D,     wl_ou, (long)D * D, stream);
        launch_cvt(fcw  + (size_t)l * MLPD * D,  wl_fc, (long)MLPD * D, stream);
        launch_cvt(cpw  + (size_t)l * D * MLPD,  wl_cp, (long)D * MLPD, stream);

        // ---- LN1 + QKV projection -----------------------------------
        ln_kernel<<<M2, 256, 0, stream>>>(h, ln1w + l * D, ln1b + l * D, an, M2, 1);
        launch_gemm(an, wl_in, inb + l * 3 * D, nullptr, qkv,
                    M2, 3 * D, D, D, D, 3 * D, 0, 0, 0, 1, 1.f, 0, stream);

        // ---- RoPE + head split (bf16 out) ---------------------------
        rope_split<<<(B * H * S * HD) / 256, 256, 0, stream>>>(qkv, rpe, Qb, Kb, Vtb);

        // ---- scores = (Q K^T) * scale, batched over heads per batch --
        for (int b = 0; b < B; b++)
            launch_gemm(Qb + (size_t)b * H * S * HD, Kb + (size_t)b * H * S * HD,
                        nullptr, nullptr, sc + (size_t)b * H * S * S,
                        S, S, HD, HD, HD, S,
                        (long)S * HD, (long)S * HD, (long)S * S,
                        H, scale, 0, stream);

        // ---- softmax; P written in place as bf16 (row stride 2*S) ----
        softmax_rows_bf16<<<B * H * S, 256, 0, stream>>>(sc);

        // ---- O = P V^T  (bf16 out) ----------------------------------
        for (int b = 0; b < B; b++)
            launch_gemm((const u16*)(sc + (size_t)b * H * S * S),
                        Vtb + (size_t)b * H * HD * S,
                        nullptr, nullptr, ob + (size_t)b * D,
                        S, HD, S, 2 * S, S, B * D,
                        2L * S * S, (long)HD * S, (long)HD,
                        H, 1.f, 4, stream);

        // ---- out projection + residual (fp32 h) ---------------------
        launch_gemm(ob, wl_ou, outb + l * D, h, h,
                    M2, D, D, D, D, D, 0, 0, 0, 1, 1.f, 2, stream);

        // ---- MLP: LN2 -> fc (GELU, bf16) -> c_proj + residual -------
        ln_kernel<<<M2, 256, 0, stream>>>(h, ln2w + l * D, ln2b + l * D, an, M2, 1);
        launch_gemm(an, wl_fc, fcb + l * MLPD, nullptr, midb,
                    M2, MLPD, D, D, D, MLPD, 0, 0, 0, 1, 1.f, 1 | 4, stream);
        launch_gemm(midb, wl_cp, cpb + l * D, h, h,
                    M2, D, MLPD, MLPD, MLPD, D, 0, 0, 0, 1, 1.f, 2, stream);
    }

    // ---- Attention pooling head -------------------------------------
    meanpool<<<8, 256, 0, stream>>>(h, xq);
    ln_kernel<<<B, 256, 0, stream>>>(xq, pnqw, pnqb, xqn, B, 0);     // fp32
    ln_kernel<<<M2, 256, 0, stream>>>(h, pnkw, pnkb, an, M2, 1);     // xk bf16
    ln_kernel<<<M2, 256, 0, stream>>>(h, pnvw, pnvb, midb, M2, 1);   // xv bf16
    launch_cvt(pkw, wl_in, (long)D * D, stream);                     // pk bf16
    launch_cvt(pvw, wl_ou, (long)D * D, stream);                     // pv bf16
    rowmat<<<(B * D + 255) / 256, 256, 0, stream>>>(xqn, pqw, nullptr, qp, B, D, D);
    launch_gemm(an,   wl_in, nullptr, nullptr, Qb, M2, D, D, D, D, D,
                0, 0, 0, 1, 1.f, 4, stream);                         // kpool
    launch_gemm(midb, wl_ou, nullptr, nullptr, Kb, M2, D, D, D, D, D,
                0, 0, 0, 1, 1.f, 4, stream);                         // vpool
    pool_attn<<<B * H, 256, 0, stream>>>(qp, Qb, Kb, op);
    rowmat<<<(B * D + 255) / 256, 256, 0, stream>>>(op, ppw, ppb,
                                                    (float*)d_out, B, D, D);
}
